// HyenaBlockLI_62895501083287
// MI455X (gfx1250) — matrix operand order repaired and checked by reading, not compiled, not run on hardware
//
#include <hip/hip_runtime.h>
#include <math.h>
#include <stdint.h>

#define LSEQ  2048
#define DDIM  1024
#define BATCH 4
#define MROWS (BATCH*LSEQ)   // 8192

typedef __attribute__((ext_vector_type(16))) __bf16 v16bf;
typedef __attribute__((ext_vector_type(8)))  float  v8f;
typedef int v4i __attribute__((vector_size(16)));
typedef __attribute__((address_space(1))) v4i* gv4i_p;
typedef __attribute__((address_space(3))) v4i* lv4i_p;

#if defined(__HIP_DEVICE_COMPILE__) && __has_builtin(__builtin_amdgcn_global_load_async_to_lds_b128)
#define ASYNC_LDS 1
#else
#define ASYNC_LDS 0
#endif

__device__ __forceinline__ unsigned short f2bf(float f) {
  unsigned x = __float_as_uint(f);
  x += 0x7fffu + ((x >> 16) & 1u);   // round-to-nearest-even
  return (unsigned short)(x >> 16);
}

__device__ __forceinline__ void copy16_g2l(const unsigned short* g, unsigned short* l) {
#if ASYNC_LDS
  __builtin_amdgcn_global_load_async_to_lds_b128(
      (gv4i_p)(uintptr_t)g,   // AS1 v4i* (numeric value == generic global addr)
      (lv4i_p)l,              // generic->AS3 addrspacecast
      0, 0);
#else
  *(uint4*)l = *(const uint4*)g;
#endif
}

__device__ __forceinline__ void wait_async() {
#if ASYNC_LDS
#if __has_builtin(__builtin_amdgcn_s_wait_asynccnt)
  __builtin_amdgcn_s_wait_asynccnt(0);
#else
  asm volatile("s_wait_asynccnt 0" ::: "memory");
#endif
#endif
}

union FragAB { v16bf v; unsigned u[8]; };

// ---------------------------------------------------------------------------
// WMMA bf16 GEMM: C = A(MxK,bf16) * W(NxK,bf16)^T + bias (+ res), f32 accum.
// Tile 128x64, BK=32, 8 waves, 32x32 per wave (2x2 WMMA frags).
// Tiles filled with GLOBAL_LOAD_ASYNC_TO_LDS_B128 when available.
// LDS row stride 40 halfs (80B): 16B-aligned rows, bank-conflict-free gathers.
// EPI: 0 = +bias, 1 = +bias + residual
// ---------------------------------------------------------------------------
template<int EPI>
__global__ __launch_bounds__(256) void gemm_bf16_wmma(
    const unsigned short* __restrict__ A, const unsigned short* __restrict__ W,
    const float* __restrict__ bias, const float* __restrict__ res,
    float* __restrict__ C, int M, int N, int K)
{
  __shared__ __align__(16) unsigned short lA[128 * 40];
  __shared__ __align__(16) unsigned short lB[64 * 40];
  const int tid  = threadIdx.x;
  const int lane = tid & 31;
  const int wave = tid >> 5;
  const int bm = blockIdx.y * 128;
  const int bn = blockIdx.x * 64;
  const int wm = (wave & 3) * 32;
  const int wn = (wave >> 2) * 32;
  const int mrow = lane & 15;
  const int kbA = (lane >> 4) * 8;    // A: lane-half K base 0/8 per 4-VGPR group
  const int kbB = (lane >> 4) * 8;    // B: lane-half K base 0/8 per 4-VGPR group

  v8f acc[2][2] = {};

  for (int k0 = 0; k0 < K; k0 += 32) {
    // A tile: 128 rows x 32 halfs = 512 16B chunks; B tile: 64 rows = 256 chunks
    #pragma unroll
    for (int i = 0; i < 2; ++i) {
      int q = tid + i * 256;
      int r = q >> 2, c = (q & 3) * 8;
      copy16_g2l(A + (size_t)(bm + r) * K + (k0 + c), &lA[r * 40 + c]);
    }
    {
      int r = tid >> 2, c = (tid & 3) * 8;
      copy16_g2l(W + (size_t)(bn + r) * K + (k0 + c), &lB[r * 40 + c]);
    }
    wait_async();
    __syncthreads();

    FragAB af[2], bfr[2];
    #pragma unroll
    for (int f = 0; f < 2; ++f) {
      int row = wm + f * 16 + mrow;
      #pragma unroll
      for (int v = 0; v < 8; ++v) {
        int kk = ((v & 4) << 2) + kbA + ((v & 3) << 1);  // ISA 16-bit A 16x32 layout
        af[f].u[v] = *(const unsigned*)&lA[row * 40 + kk];
      }
      int col = wn + f * 16 + mrow;
      #pragma unroll
      for (int v = 0; v < 8; ++v) {
        int kk = ((v & 4) << 2) + kbB + ((v & 3) << 1);  // B 32x16 layout
        bfr[f].u[v] = *(const unsigned*)&lB[col * 40 + kk];
      }
    }
    #pragma unroll
    for (int i = 0; i < 2; ++i)
      #pragma unroll
      for (int j = 0; j < 2; ++j)
        acc[i][j] = __builtin_amdgcn_wmma_f32_16x16x32_bf16(
            false, af[i].v, false, bfr[j].v, (short)0, acc[i][j], false, false);
    __syncthreads();
  }

  #pragma unroll
  for (int i = 0; i < 2; ++i) {
    int rbase = bm + wm + i * 16 + (lane >> 4) * 8;   // C layout: M = r + 8*(lane>=16)
    #pragma unroll
    for (int j = 0; j < 2; ++j) {
      int col = bn + wn + j * 16 + mrow;
      float bcol = bias[col];
      #pragma unroll
      for (int r = 0; r < 8; ++r) {
        size_t idx = (size_t)(rbase + r) * N + col;
        float v = acc[i][j][r] + bcol;
        if (EPI == 1) v += res[idx];
        C[idx] = v;
      }
    }
  }
}

// ---------------------------------------------------------------------------
__global__ __launch_bounds__(256) void to_bf16_k(
    const float* __restrict__ in, unsigned short* __restrict__ out, int n)
{
  int i = blockIdx.x * 256 + threadIdx.x;
  if (i < n) out[i] = f2bf(in[i]);
}

// RMSNorm, bf16 output (feeds WMMA GEMM)
__global__ __launch_bounds__(256) void rmsnorm_bf16_k(
    const float* __restrict__ x, const float* __restrict__ w,
    unsigned short* __restrict__ o, int D)
{
  int row = blockIdx.x;
  const float* xr = x + (size_t)row * D;
  float s = 0.f;
  for (int i = threadIdx.x; i < D; i += 256) { float v = xr[i]; s += v * v; }
  __shared__ float red[256];
  red[threadIdx.x] = s;
  __syncthreads();
  for (int off = 128; off > 0; off >>= 1) {
    if (threadIdx.x < off) red[threadIdx.x] += red[threadIdx.x + off];
    __syncthreads();
  }
  float scale = rsqrtf(red[0] / (float)D + 1e-8f);
  for (int i = threadIdx.x; i < D; i += 256)
    o[(size_t)row * D + i] = f2bf(xr[i] * scale * w[i]);
}

// ---------------------------------------------------------------------------
// Depthwise causal conv (k=3) on u (B,L,3D) + split + gate-prep, transposed
// writes: x0c (B,D,L), vin = x1c*vc (B,D,L). 32x32 LDS-tile transpose.
// ---------------------------------------------------------------------------
__global__ __launch_bounds__(256) void shortconv_k(
    const float* __restrict__ u, const float* __restrict__ sfw,
    const float* __restrict__ sfb,
    float* __restrict__ x0c, float* __restrict__ vin)
{
  __shared__ float t0[32][33];
  __shared__ float t1[32][33];
  int b  = blockIdx.z;
  int l0 = blockIdx.x * 32;
  int d0 = blockIdx.y * 32;
  int tx = threadIdx.x, ty = threadIdx.y;
  const int C3 = 3 * DDIM;
  #pragma unroll
  for (int sIt = 0; sIt < 4; ++sIt) {
    int li = ty + sIt * 8;
    int l  = l0 + li;
    int d  = d0 + tx;
    float vals[3];
    #pragma unroll
    for (int p = 0; p < 3; ++p) {
      int c = d + p * DDIM;
      const float* ub = u + ((size_t)b * LSEQ) * C3 + c;
      float w0 = sfw[c * 3 + 0], w1 = sfw[c * 3 + 1], w2 = sfw[c * 3 + 2];
      float a = w2 * ub[(size_t)l * C3];
      if (l >= 1) a += w1 * ub[(size_t)(l - 1) * C3];
      if (l >= 2) a += w0 * ub[(size_t)(l - 2) * C3];
      vals[p] = a + sfb[c];
    }
    t0[li][tx] = vals[0];
    t1[li][tx] = vals[1] * vals[2];
  }
  __syncthreads();
  #pragma unroll
  for (int sIt = 0; sIt < 4; ++sIt) {
    int dl = ty + sIt * 8;
    size_t idx = ((size_t)(b * DDIM + d0 + dl)) * LSEQ + l0 + tx;
    x0c[idx] = t0[tx][dl];
    vin[idx] = t1[tx][dl];
  }
}

// ---------------------------------------------------------------------------
// Implicit filter MLP: one 64-thread block per time position
// ---------------------------------------------------------------------------
__global__ __launch_bounds__(64) void implicit_filter_k(
    const float* __restrict__ w1, const float* __restrict__ b1, const float* __restrict__ f1,
    const float* __restrict__ w2, const float* __restrict__ b2, const float* __restrict__ f2,
    const float* __restrict__ w3, float* __restrict__ kout)
{
  __shared__ float h1s[64], h2s[64];
  int l = blockIdx.x, j = threadIdx.x;
  float t   = (float)l / (float)(LSEQ - 1);
  float ang = 1e-4f * 6.283185307179586f * (float)l / (float)LSEQ;
  float z1  = cosf(ang), z2 = -sinf(ang);
  float a = w1[j * 3 + 0] * t + w1[j * 3 + 1] * z1 + w1[j * 3 + 2] * z2 + b1[j];
  h1s[j] = sinf(f1[j] * a);
  __syncthreads();
  float sacc = b2[j];
  for (int m = 0; m < 64; ++m) sacc += w2[j * 64 + m] * h1s[m];
  h2s[j] = sinf(f2[j] * sacc);
  __syncthreads();
  const float mind = -4.605170185988091f / 1.5f;   // log(1e-2)/slow
  const float maxd = -4.605170185988091f / 0.3f;   // log(1e-2)/fast
  for (int i = 0; i < 16; ++i) {
    int d = j + i * 64;
    float acc = 0.f;
    for (int m = 0; m < 64; ++m) acc += w3[d * 64 + m] * h2s[m];
    float delta = mind + (maxd - mind) * (float)d / (float)(DDIM - 1);
    kout[(size_t)d * LSEQ + l] = acc * expf(-t * fabsf(delta));
  }
}

// ---------------------------------------------------------------------------
// 4096-point radix-2 DIT FFT in LDS (256 threads). tw[k] = exp(-2pi i k/4096).
// conjtw=true => unscaled inverse transform.
// ---------------------------------------------------------------------------
__device__ __forceinline__ void fft4096(float2* s, const float2* tw, int tid, bool conjtw) {
  for (int i = tid; i < 4096; i += 256) {
    int j = __brev((unsigned)i) >> 20;
    if (j > i) { float2 t = s[i]; s[i] = s[j]; s[j] = t; }
  }
  __syncthreads();
  for (int st = 1; st <= 12; ++st) {
    int half = 1 << (st - 1);
    for (int j = tid; j < 2048; j += 256) {
      int p  = j & (half - 1);
      int g  = j >> (st - 1);
      int i0 = (g << st) + p;
      int i1 = i0 + half;
      float2 w = tw[p << (12 - st)];
      float wy = conjtw ? -w.y : w.y;
      float2 bb = s[i1], aa = s[i0];
      float txr = bb.x * w.x - bb.y * wy;
      float tyi = bb.x * wy + bb.y * w.x;
      s[i0] = make_float2(aa.x + txr, aa.y + tyi);
      s[i1] = make_float2(aa.x - txr, aa.y - tyi);
    }
    __syncthreads();
  }
}

__device__ __forceinline__ void fill_tw(float2* tw, int tid) {
  for (int i = tid; i < 2048; i += 256) {
    float ang = -6.283185307179586f * (float)i / 4096.0f;
    float sv, cv; sincosf(ang, &sv, &cv);
    tw[i] = make_float2(cv, sv);
  }
}

// Khat = FFT(zero-pad(k_row)) / 4096, one block per channel d
__global__ __launch_bounds__(256) void fft_filter_k(
    const float* __restrict__ kf, float2* __restrict__ Kf)
{
  __shared__ float2 s[4096];
  __shared__ float2 tw[2048];
  int d = blockIdx.x, tid = threadIdx.x;
  fill_tw(tw, tid);
  for (int i = tid; i < 4096; i += 256)
    s[i] = make_float2(i < LSEQ ? kf[(size_t)d * LSEQ + i] : 0.f, 0.f);
  __syncthreads();
  fft4096(s, tw, tid, false);
  const float sc = 1.0f / 4096.0f;
  for (int i = tid; i < 4096; i += 256)
    Kf[(size_t)d * 4096 + i] = make_float2(s[i].x * sc, s[i].y * sc);
}

// y = IDFT_noscale(FFT(vin)*Khat)[0:L] + vin*bias  (in place on vin row)
// inverse via conj(FFT(conj(.))) ; one block per (b,d) row
__global__ __launch_bounds__(256) void fftconv_k(
    float* __restrict__ vio, const float2* __restrict__ Kf,
    const float* __restrict__ bias)
{
  __shared__ float2 s[4096];
  __shared__ float2 tw[2048];
  int bd = blockIdx.x, tid = threadIdx.x;
  int d  = bd & (DDIM - 1);
  fill_tw(tw, tid);
  for (int i = tid; i < 4096; i += 256)
    s[i] = make_float2(i < LSEQ ? vio[(size_t)bd * LSEQ + i] : 0.f, 0.f);
  __syncthreads();
  fft4096(s, tw, tid, false);
  for (int i = tid; i < 4096; i += 256) {
    float2 v = s[i], k = Kf[(size_t)d * 4096 + i];
    s[i] = make_float2(v.x * k.x - v.y * k.y, -(v.x * k.y + v.y * k.x)); // conj(V*K)
  }
  __syncthreads();
  fft4096(s, tw, tid, false);
  for (int i = tid; i < LSEQ; i += 256) {
    size_t idx = (size_t)bd * LSEQ + i;
    float vv = vio[idx];
    vio[idx] = s[i].x + vv * bias[d];   // Re(conj(x)) == x.x
  }
}

// yg(B,L,D) bf16 = y(B,D,L) * x0c(B,D,L), transposed via 32x32 LDS tile
__global__ __launch_bounds__(256) void gate_transpose_bf16_k(
    const float* __restrict__ y, const float* __restrict__ x0,
    unsigned short* __restrict__ yg)
{
  __shared__ float t[32][33];
  int b  = blockIdx.z;
  int l0 = blockIdx.x * 32;
  int d0 = blockIdx.y * 32;
  int tx = threadIdx.x, ty = threadIdx.y;
  #pragma unroll
  for (int sIt = 0; sIt < 4; ++sIt) {
    int dl = ty + sIt * 8;
    size_t idx = ((size_t)(b * DDIM + d0 + dl)) * LSEQ + l0 + tx;
    t[tx][dl] = y[idx] * x0[idx];
  }
  __syncthreads();
  #pragma unroll
  for (int sIt = 0; sIt < 4; ++sIt) {
    int ll = ty + sIt * 8;
    yg[((size_t)b * LSEQ + l0 + ll) * DDIM + d0 + tx] = f2bf(t[ll][tx]);
  }
}

// gelu (tanh approx) f32 -> bf16
__global__ __launch_bounds__(256) void gelu_bf16_k(
    const float* __restrict__ g, unsigned short* __restrict__ gb, long long n)
{
  long long i = (long long)blockIdx.x * 256 + threadIdx.x;
  if (i < n) {
    float v = g[i];
    float t = tanhf(0.7978845608028654f * (v + 0.044715f * v * v * v));
    gb[i] = f2bf(0.5f * v * (1.f + t));
  }
}

// ---------------------------------------------------------------------------
extern "C" void kernel_launch(void* const* d_in, const int* in_sizes, int n_in,
                              void* d_out, int out_size, void* d_ws, size_t ws_size,
                              hipStream_t stream) {
  (void)in_sizes; (void)n_in; (void)out_size; (void)ws_size;
  const float* x     = (const float*)d_in[0];
  const float* nin_w = (const float*)d_in[1];
  const float* ipw   = (const float*)d_in[2];
  const float* ipb   = (const float*)d_in[3];
  const float* sfw   = (const float*)d_in[4];
  const float* sfb   = (const float*)d_in[5];
  const float* mw1   = (const float*)d_in[6];
  const float* mb1   = (const float*)d_in[7];
  const float* fr1   = (const float*)d_in[8];
  const float* mw2   = (const float*)d_in[9];
  const float* mb2   = (const float*)d_in[10];
  const float* fr2   = (const float*)d_in[11];
  const float* mw3   = (const float*)d_in[12];
  const float* fbias = (const float*)d_in[13];
  const float* opw   = (const float*)d_in[14];
  const float* opb   = (const float*)d_in[15];
  const float* nw    = (const float*)d_in[16];
  const float* f1w   = (const float*)d_in[17];
  const float* f1b   = (const float*)d_in[18];
  const float* f2w   = (const float*)d_in[19];
  const float* f2b   = (const float*)d_in[20];
  float* out = (float*)d_out;

  float* ws = (float*)d_ws;
  const size_t MD = (size_t)MROWS * DDIM;            // 8M elements
  float* u    = ws;                                   // 3*MD f32 (B,L,3D)
  float* x0c  = u + 3 * MD;                           // MD f32 (B,D,L)
  float* vin  = x0c + MD;                             // MD f32 (B,D,L)
  float* kft  = vin + MD;                             // D*L f32
  float* Kf   = kft + (size_t)DDIM * LSEQ;            // D*4096*2 f32 (complex)
  unsigned short* xnb  = (unsigned short*)(Kf + (size_t)DDIM * 4096 * 2); // MD bf16
  unsigned short* ipwb = xnb + MD;                    // 3M bf16
  unsigned short* opwb = ipwb + 3 * (size_t)DDIM * DDIM;
  unsigned short* f1wb = opwb + (size_t)DDIM * DDIM;
  unsigned short* f2wb = f1wb + 2 * (size_t)DDIM * DDIM;
  // reuse (stream-ordered): yg/gb overlay Kf region after fftconv is done,
  // h/g overlay u after shortconv is done, hnb overlays xnb after in_proj.
  unsigned short* yg  = (unsigned short*)Kf;          // MD bf16
  unsigned short* gb  = (unsigned short*)Kf;          // 2*MD bf16 (after yg dead)
  float* h   = u;                                     // MD f32
  float* g   = u + MD;                                // 2*MD f32
  unsigned short* hnb = xnb;                          // MD bf16

  dim3 b256(256);
  const int DD = DDIM * DDIM;

  // one-time (per launch) weight conversions to bf16
  to_bf16_k<<<(3 * DD + 255) / 256, b256, 0, stream>>>(ipw, ipwb, 3 * DD);
  to_bf16_k<<<(DD + 255) / 256, b256, 0, stream>>>(opw, opwb, DD);
  to_bf16_k<<<(2 * DD + 255) / 256, b256, 0, stream>>>(f1w, f1wb, 2 * DD);
  to_bf16_k<<<(2 * DD + 255) / 256, b256, 0, stream>>>(f2w, f2wb, 2 * DD);

  rmsnorm_bf16_k<<<MROWS, b256, 0, stream>>>(x, nin_w, xnb, DDIM);

  gemm_bf16_wmma<0><<<dim3(3 * DDIM / 64, MROWS / 128), b256, 0, stream>>>(
      xnb, ipwb, ipb, nullptr, u, MROWS, 3 * DDIM, DDIM);

  shortconv_k<<<dim3(LSEQ / 32, DDIM / 32, BATCH), dim3(32, 8), 0, stream>>>(
      u, sfw, sfb, x0c, vin);

  implicit_filter_k<<<LSEQ, 64, 0, stream>>>(mw1, mb1, fr1, mw2, mb2, fr2, mw3, kft);

  fft_filter_k<<<DDIM, b256, 0, stream>>>(kft, (float2*)Kf);

  fftconv_k<<<BATCH * DDIM, b256, 0, stream>>>(vin, (const float2*)Kf, fbias);

  gate_transpose_bf16_k<<<dim3(LSEQ / 32, DDIM / 32, BATCH), dim3(32, 8), 0, stream>>>(
      vin, x0c, yg);

  gemm_bf16_wmma<1><<<dim3(DDIM / 64, MROWS / 128), b256, 0, stream>>>(
      yg, opwb, opb, x, h, MROWS, DDIM, DDIM);

  rmsnorm_bf16_k<<<MROWS, b256, 0, stream>>>(h, nw, hnb, DDIM);

  gemm_bf16_wmma<0><<<dim3(2 * DDIM / 64, MROWS / 128), b256, 0, stream>>>(
      hnb, f1wb, f1b, nullptr, g, MROWS, 2 * DDIM, DDIM);

  long long ngel = (long long)MROWS * 2 * DDIM;
  gelu_bf16_k<<<(unsigned)((ngel + 255) / 256), b256, 0, stream>>>(g, gb, ngel);

  gemm_bf16_wmma<1><<<dim3(DDIM / 64, MROWS / 128), b256, 0, stream>>>(
      gb, f2wb, f2b, h, out, MROWS, DDIM, 2 * DDIM);
}
